// SAGNN_82867099009184
// MI455X (gfx1250) — matrix operand order repaired and checked by reading, not compiled, not run on hardware
//
#include <hip/hip_runtime.h>
#include <hip/hip_bf16.h>

typedef __attribute__((ext_vector_type(16))) _Float16 v16h;
typedef __attribute__((ext_vector_type(8)))  _Float16 h8;
typedef __attribute__((ext_vector_type(2)))  _Float16 h2;
typedef __attribute__((ext_vector_type(8)))  float    v8f;

#define TPB 256

// ---------------------------------------------------------------------------
// Strip-tiled WMMA GEMM on pre-converted f16 operands.
//   A  : f16 [M, Kp] row-major (Kp multiple of 32, zero padded)
//   Wt : f16 [NT*16, Kp] row-major == W^T, zero padded
//   C  : f32 or f16 [M, NT*16]
// One wave32 owns a 16-row strip across ALL NT column tiles: A fragments are
// read once per K-step (instead of NT times), B tiles stream from L2/WGP$.
// M must be a multiple of 16 (true for all call sites).
// Fragment layouts per CDNA5 ISA 7.12.2:
//   A lane L: row=L&15, kh=L>>4; elems 0..7 -> K=kh*8+e ; 8..15 -> K=16+kh*8+e
//   B lane L: col=L&15, kh=L>>4; elems 0..7 -> K=kh*8+e ; 8..15 -> K=16+kh*8+e
//   C lane L: col=L&15; vgpr r -> row=(L>>4)*8+r
// ---------------------------------------------------------------------------
template <int NT>
__global__ void k_wmma_strip(const _Float16* __restrict__ A,
                             const _Float16* __restrict__ Wt,
                             const float* __restrict__ bias,
                             void* __restrict__ C,
                             int M, int Kp, int act, int outHalf)
{
    const int lane = threadIdx.x & 31;
    const int wave = threadIdx.x >> 5;
    const int tm   = blockIdx.x * 8 + wave;            // wave-uniform strip id
    if (tm >= (M >> 4)) return;                        // uniform: EXEC stays all-1
    const int rla = lane & 15;
    const int kh  = lane >> 4;

    const h8* __restrict__ Ap = (const h8*)(A  + (size_t)(tm * 16 + rla) * Kp);
    const h8* __restrict__ Bp = (const h8*)(Wt + (size_t)rla * Kp);
    const size_t tstr = (size_t)2 * Kp;                // 16 rows * Kp / 8 per h8

    v8f acc[NT];
#pragma unroll
    for (int t = 0; t < NT; ++t) acc[t] = (v8f){};

    const int steps = Kp >> 5;
    for (int s = 0; s < steps; ++s) {
        h8 a0 = Ap[s * 4 + kh];
        h8 a1 = Ap[s * 4 + 2 + kh];
        v16h av = __builtin_shufflevector(a0, a1, 0,1,2,3,4,5,6,7,8,9,10,11,12,13,14,15);
#pragma unroll
        for (int t = 0; t < NT; ++t) {
            const h8* Bt = Bp + (size_t)t * tstr;
            h8 b0 = Bt[s * 4 + kh];
            h8 b1 = Bt[s * 4 + 2 + kh];
            v16h bv = __builtin_shufflevector(b0, b1, 0,1,2,3,4,5,6,7,8,9,10,11,12,13,14,15);
            acc[t] = __builtin_amdgcn_wmma_f32_16x16x32_f16(false, av, false, bv,
                                                            (short)0, acc[t], false, false);
        }
    }

    const int Nc = NT * 16;
#pragma unroll
    for (int t = 0; t < NT; ++t) {
        const int col = t * 16 + rla;
        const float bb = bias ? bias[col] : 0.0f;
#pragma unroll
        for (int r = 0; r < 8; ++r) {
            int orow = tm * 16 + kh * 8 + r;
            float v = acc[t][r] + bb;
            if (act == 1) v = fmaxf(v, 0.0f);
            if (outHalf) ((_Float16*)C)[(size_t)orow * Nc + col] = (_Float16)v;
            else         ((float*)C)[(size_t)orow * Nc + col] = v;
        }
    }
}

// ---------------------------------------------------------------------------
// f32 -> f16 conversion / layout kernels
// ---------------------------------------------------------------------------
__global__ void k_cvt_a(const float* __restrict__ src, _Float16* __restrict__ dst,
                        long n, int K, int Kp) {
    long i = (long)blockIdx.x * blockDim.x + threadIdx.x;
    if (i >= n) return;
    long r = i / Kp;
    int  k = (int)(i - r * Kp);
    dst[i] = (k < K) ? (_Float16)src[r * K + k] : (_Float16)0.0f;
}

// Wt[N,Kp] f16 = pad(transpose(w[K,N] f32))
__global__ void k_cvt_wt(const float* __restrict__ w, _Float16* __restrict__ wt,
                         long n, int K, int N, int Kp) {
    long i = (long)blockIdx.x * blockDim.x + threadIdx.x;
    if (i >= n) return;
    long c = i / Kp;
    int  k = (int)(i - c * Kp);
    wt[i] = (k < K) ? (_Float16)w[(long)k * N + c] : (_Float16)0.0f;
}

__global__ void k_cvt_h(const float* __restrict__ s, _Float16* __restrict__ d, long n) {
    long i = (long)blockIdx.x * blockDim.x + threadIdx.x;
    if (i < n) d[i] = (_Float16)s[i];
}

// concat: dst[rows,dstCols] f16 at column colOff <- src[rows,srcCols] f32
__global__ void k_concat_cols_h(_Float16* __restrict__ dst, int dstCols, int colOff,
                                const float* __restrict__ src, int srcCols, long n) {
    long i = (long)blockIdx.x * blockDim.x + threadIdx.x;
    if (i >= n) return;
    long r = i / srcCols;
    int  c = (int)(i - r * srcCols);
    dst[r * dstCols + colOff + c] = (_Float16)src[i];
}

// ---------------------------------------------------------------------------
// Elementwise / reduction helpers (hidden width 128 where noted)
// ---------------------------------------------------------------------------
__global__ void k_fill(float* __restrict__ p, float v, long n) {
    long i = (long)blockIdx.x * blockDim.x + threadIdx.x;
    if (i < n) p[i] = v;
}

__global__ void k_copy(float* __restrict__ d, const float* __restrict__ s, long n) {
    long i = (long)blockIdx.x * blockDim.x + threadIdx.x;
    if (i < n) d[i] = s[i];
}

// y_h = (f16)(a + b)
__global__ void k_add_h(const float* __restrict__ a, const float* __restrict__ b,
                        _Float16* __restrict__ y, long n) {
    long i = (long)blockIdx.x * blockDim.x + threadIdx.x;
    if (i < n) y[i] = (_Float16)(a[i] + b[i]);
}

__global__ void k_scatter_rows(float* __restrict__ dst, const float* __restrict__ src,
                               const int* __restrict__ idx, long n, int cols) {
    long i = (long)blockIdx.x * blockDim.x + threadIdx.x;
    if (i >= n) return;
    long r = i / cols;
    int  c = (int)(i - r * cols);
    dst[(long)idx[r] * cols + c] = src[i];
}

__global__ void k_segsum(float* __restrict__ acc, float* __restrict__ cnt,
                         const float* __restrict__ src, const int* __restrict__ seg,
                         long n, int cols) {
    long i = (long)blockIdx.x * blockDim.x + threadIdx.x;
    if (i >= n) return;
    long r = i / cols;
    int  c = (int)(i - r * cols);
    int  s = seg[r];
    atomicAdd(&acc[(long)s * cols + c], src[i]);
    if (c == 0) atomicAdd(&cnt[s], 1.0f);
}

__global__ void k_div_cnt(float* __restrict__ acc, const float* __restrict__ cnt,
                          long n, int cols) {
    long i = (long)blockIdx.x * blockDim.x + threadIdx.x;
    if (i >= n) return;
    long r = i / cols;
    acc[i] /= fmaxf(cnt[r], 1.0f);
}

__global__ void k_colstats(const float* __restrict__ x, long M,
                           float* __restrict__ sum, float* __restrict__ sumsq) {
    int  c      = threadIdx.x & 127;
    long r      = ((long)blockIdx.x * blockDim.x + threadIdx.x) >> 7;
    long stride = ((long)gridDim.x * blockDim.x) >> 7;
    float s = 0.0f, s2 = 0.0f;
    for (; r < M; r += stride) {
        float v = x[r * 128 + c];
        s += v; s2 += v * v;
    }
    atomicAdd(&sum[c], s);
    atomicAdd(&sumsq[c], s2);
}

__global__ void k_bn_finalize(const float* __restrict__ sum, const float* __restrict__ sumsq,
                              const float* __restrict__ g, const float* __restrict__ b,
                              float invM, float* __restrict__ scale, float* __restrict__ shift) {
    int c = threadIdx.x;
    if (c >= 128) return;
    float m  = sum[c] * invM;
    float v  = sumsq[c] * invM - m * m;
    float sc = g[c] * rsqrtf(v + 1e-5f);
    scale[c] = sc;
    shift[c] = b[c] - m * sc;
}

__global__ void k_bn_relu(const float* __restrict__ x, const float* __restrict__ sc,
                          const float* __restrict__ sh, float* __restrict__ y, long n) {
    long i = (long)blockIdx.x * blockDim.x + threadIdx.x;
    if (i >= n) return;
    int c = (int)(i & 127);
    y[i] = fmaxf(x[i] * sc[c] + sh[c], 0.0f);
}

__global__ void k_bn_relu_h(const float* __restrict__ x, const float* __restrict__ sc,
                            const float* __restrict__ sh, _Float16* __restrict__ y, long n) {
    long i = (long)blockIdx.x * blockDim.x + threadIdx.x;
    if (i >= n) return;
    int c = (int)(i & 127);
    y[i] = (_Float16)fmaxf(x[i] * sc[c] + sh[c], 0.0f);
}

// y = relu(bn(x)) + res ; res = y
__global__ void k_bn_relu_res(const float* __restrict__ x, const float* __restrict__ sc,
                              const float* __restrict__ sh, float* __restrict__ res,
                              float* __restrict__ y, long n) {
    long i = (long)blockIdx.x * blockDim.x + threadIdx.x;
    if (i >= n) return;
    int c = (int)(i & 127);
    float v = fmaxf(x[i] * sc[c] + sh[c], 0.0f) + res[i];
    y[i]   = v;
    res[i] = v;
}

// GINE message + scatter, 2 columns/thread:
//   agg[dst[j],c] += relu(x[src[j],c] + e[j,c])  for c = 2t, 2t+1
__global__ void k_edge_msg(const _Float16* __restrict__ x, const _Float16* __restrict__ e,
                           const int* __restrict__ src, const int* __restrict__ dst,
                           float* __restrict__ agg, long n2, long nE) {
    long i = (long)blockIdx.x * blockDim.x + threadIdx.x;
    if (i >= n2) return;
    long j  = i >> 6;
    int  c2 = (int)(i & 63) * 2;
    int  s  = src[j];
    int  d  = dst[j];
    // prefetch the gather row of an edge ahead (data-dependent address)
    if (c2 == 0 && j + 8 < nE)
        __builtin_prefetch(x + (long)src[j + 8] * 128, 0, 1);
    h2 xv = *(const h2*)(x + (long)s * 128 + c2);
    h2 ev = *(const h2*)(e + j * 128 + c2);
    float m0 = fmaxf((float)xv[0] + (float)ev[0], 0.0f);
    float m1 = fmaxf((float)xv[1] + (float)ev[1], 0.0f);
    atomicAdd(&agg[(long)d * 128 + c2],     m0);
    atomicAdd(&agg[(long)d * 128 + c2 + 1], m1);
}

// out[g] = sum_c d1[g,c]*w[c] + b[0]
__global__ void k_dec2(const float* __restrict__ d1, const float* __restrict__ w,
                       const float* __restrict__ b, float* __restrict__ out) {
    __shared__ float red[128];
    int r = blockIdx.x, t = threadIdx.x;
    red[t] = d1[(long)r * 128 + t] * w[t];
    __syncthreads();
    for (int s = 64; s > 0; s >>= 1) {
        if (t < s) red[t] += red[t + s];
        __syncthreads();
    }
    if (t == 0) out[r] = red[0] + b[0];
}

// ---------------------------------------------------------------------------
// Host orchestration
// ---------------------------------------------------------------------------
static inline dim3 g1(long n) { return dim3((unsigned)((n + TPB - 1) / TPB)); }

extern "C" void kernel_launch(void* const* d_in, const int* in_sizes, int n_in,
                              void* d_out, int out_size, void* d_ws, size_t ws_size,
                              hipStream_t stream) {
    const long N    = in_sizes[0] / 64;    // 50000
    const long E    = in_sizes[1] / 8;     // 800000
    const long NSUB = in_sizes[2] / 8;     // 400000
    const long G    = out_size;            // 256

    const float* x         = (const float*)d_in[0];
    const float* edge_attr = (const float*)d_in[1];
    const float* ego_RWPE  = (const float*)d_in[2];
    const float* glo_RWPE  = (const float*)d_in[3];
    const float* cut_RWPE  = (const float*)d_in[4];
    const int*   e_src     = (const int*)d_in[5];
    const int*   e_dst     = ((const int*)d_in[5]) + E;
    const int*   sxi       = (const int*)d_in[6];
    const int*   sb        = (const int*)d_in[7];
    const int*   batch     = (const int*)d_in[8];

    auto P = [&](int i) { return (const float*)d_in[i]; };
    int pi = 9;
    const float *cut_emb_w = P(pi++), *cut_emb_b = P(pi++);
    const float *cut_out_w = P(pi++), *cut_out_b = P(pi++);
    const float *dec1_w    = P(pi++), *dec1_b    = P(pi++);
    const float *dec2_w    = P(pi++), *dec2_b    = P(pi++);
    const float *ego_emb_w = P(pi++), *ego_emb_b = P(pi++);
    const float *ego_out_w = P(pi++), *ego_out_b = P(pi++);
    const float *enc_w     = P(pi++), *enc_b     = P(pi++);
    const float *enc_bn_g  = P(pi++), *enc_bn_b  = P(pi++);
    const float *glo_emb_w = P(pi++), *glo_emb_b = P(pi++);
    const float *glo_out_w = P(pi++), *glo_out_b = P(pi++);
    const float *Lbn0g[2], *Lbn0b[2], *Lbn1g[2], *Lbn1b[2];
    const float *Leew[2], *Leeb[2], *Leebng[2], *Leebnb[2];
    const float *Lg1a[2], *Lg1b[2], *Lg2a[2], *Lg2b[2];
    for (int l = 0; l < 2; ++l) {
        Lbn0g[l] = P(pi++); Lbn0b[l] = P(pi++);
        Lbn1g[l] = P(pi++); Lbn1b[l] = P(pi++);
        Leew[l]  = P(pi++); Leeb[l]  = P(pi++);
        Leebng[l]= P(pi++); Leebnb[l]= P(pi++);
        Lg1a[l]  = P(pi++); Lg1b[l]  = P(pi++);
        Lg2a[l]  = P(pi++); Lg2b[l]  = P(pi++);
    }
    const float *merge_w  = P(pi++), *merge_b  = P(pi++);
    const float *merge2_w = P(pi++), *merge2_b = P(pi++);
    const float *whp_w    = P(pi++), *whp_b    = P(pi++);
    const float *whp2_w   = P(pi++), *whp2_b   = P(pi++);
    (void)n_in; (void)ws_size;

    // workspace bump allocator
    char*  base = (char*)d_ws;
    size_t off  = 0;
    auto allocB = [&](size_t bytes) -> void* {
        void* p = (void*)(base + off);
        off = (off + bytes + 255) & ~(size_t)255;
        return p;
    };
    const size_t SZ_N128 = (size_t)N * 128 * 4;
    float* EBUF = (float*)allocB((size_t)E * 128 * 4);
    float* H    = (float*)allocB(SZ_N128);
    float* GLO  = (float*)allocB(SZ_N128);
    float* CT   = (float*)allocB(SZ_N128);
    float* EP   = (float*)allocB(SZ_N128);
    float* XE   = (float*)allocB(SZ_N128);
    float* XC   = (float*)allocB(SZ_N128);
    float* PEb  = (float*)allocB(SZ_N128);
    float* PCb  = (float*)allocB(SZ_N128);
    float* AGG  = (float*)allocB(SZ_N128);
    float* TMP  = (float*)allocB(SZ_N128);
    _Float16* EH = (_Float16*)allocB((size_t)E * 128 * 2);
    _Float16* XH = (_Float16*)allocB((size_t)N * 128 * 2);
    _Float16* AH = (_Float16*)allocB((size_t)E * 32 * 2);
    _Float16* BH = (_Float16*)allocB((size_t)N * 128 * 2);
    _Float16* WT = (_Float16*)allocB((size_t)384 * 128 * 2);
    float* CNT  = (float*)allocB((size_t)N * 4);
    float* CNTG = (float*)allocB((size_t)G * 4);
    float* SUM  = (float*)allocB(128 * 4);
    float* SSQ  = (float*)allocB(128 * 4);
    float* SCALE= (float*)allocB(128 * 4);
    float* SHIFT= (float*)allocB(128 * 4);
    float* R    = (float*)allocB((size_t)G * 128 * 4);
    float* RC   = (float*)allocB((size_t)G * 128 * 4);
    float* D1   = (float*)allocB((size_t)G * 128 * 4);
    float* EO   = (float*)allocB((size_t)N * 16 * 4);
    float* CO   = (float*)allocB((size_t)N * 16 * 4);
    float* GO   = (float*)allocB((size_t)N * 16 * 4);

    auto cvtA = [&](const float* s, long M, int K, int Kp) {
        k_cvt_a<<<g1(M * Kp), TPB, 0, stream>>>(s, AH, M * Kp, K, Kp);
    };
    auto cvtW = [&](const float* w, int K, int Nc, int Kp) {
        k_cvt_wt<<<g1((long)Nc * Kp), TPB, 0, stream>>>(w, WT, (long)Nc * Kp, K, Nc, Kp);
    };
    auto gemm = [&](const _Float16* A, const float* bias, void* C,
                    long M, int Kp, int Nc, int act, int outHalf) {
        dim3 grid((unsigned)((M / 16 + 7) / 8));
        if (Nc == 128)
            k_wmma_strip<8><<<grid, TPB, 0, stream>>>(A, WT, bias, C, (int)M, Kp, act, outHalf);
        else
            k_wmma_strip<1><<<grid, TPB, 0, stream>>>(A, WT, bias, C, (int)M, Kp, act, outHalf);
    };
    auto zero = [&](float* p, long n) {
        k_fill<<<g1(n), TPB, 0, stream>>>(p, 0.0f, n);
    };
    auto bn_stats = [&](const float* src, long M) {
        zero(SUM, 128); zero(SSQ, 128);
        k_colstats<<<dim3(512), TPB, 0, stream>>>(src, M, SUM, SSQ);
    };
    auto bn_fin = [&](const float* g, const float* b, long M) {
        k_bn_finalize<<<dim3(1), dim3(128), 0, stream>>>(SUM, SSQ, g, b,
                                                         1.0f / (float)M, SCALE, SHIFT);
    };
    auto concatH = [&](int dstCols, int colOff, const float* src, int srcCols, long rows) {
        k_concat_cols_h<<<g1(rows * srcCols), TPB, 0, stream>>>(AH, dstCols, colOff,
                                                                src, srcCols, rows * srcCols);
    };

    const long n128 = N * 128, e128 = E * 128;

    // h = relu(bn(x @ enc))
    cvtA(x, N, 64, 64); cvtW(enc_w, 64, 128, 64);
    gemm(AH, enc_b, TMP, N, 64, 128, 0, 0);
    bn_stats(TMP, N); bn_fin(enc_bn_g, enc_bn_b, N);
    k_bn_relu<<<g1(n128), TPB, 0, stream>>>(TMP, SCALE, SHIFT, H, n128);

    // embeddings (K padded 8/16 -> 32)
    cvtA(ego_RWPE, NSUB, 8, 32); cvtW(ego_emb_w, 8, 128, 32);
    gemm(AH, ego_emb_b, EBUF, NSUB, 32, 128, 0, 0);            // EGO in EBUF arena
    cvtA(glo_RWPE, N, 16, 32); cvtW(glo_emb_w, 16, 128, 32);
    gemm(AH, glo_emb_b, GLO, N, 32, 128, 0, 0);
    cvtA(cut_RWPE, N, 8, 32); cvtW(cut_emb_w, 8, 128, 32);
    gemm(AH, cut_emb_b, TMP, N, 32, 128, 0, 0);                // CUT in TMP

    // cut_trans scatter (permutation)
    zero(CT, n128);
    k_scatter_rows<<<g1(n128), TPB, 0, stream>>>(CT, TMP, sxi, n128, 128);

    // ego_pooled = seg_mean(EGO, subgraphs_batch, N)
    zero(EP, n128); zero(CNT, N);
    k_segsum<<<g1(NSUB * 128), TPB, 0, stream>>>(EP, CNT, EBUF, sb, NSUB * 128, 128);
    k_div_cnt<<<g1(n128), TPB, 0, stream>>>(EP, CNT, n128, 128);

    // merges
    concatH(384, 0, H, 128, N);
    concatH(384, 128, EP, 128, N);
    concatH(384, 256, GLO, 128, N);
    cvtW(merge_w, 384, 128, 384);
    gemm(AH, merge_b, XE, N, 384, 128, 0, 0);
    concatH(384, 128, CT, 128, N);
    cvtW(merge2_w, 384, 128, 384);
    gemm(AH, merge2_b, XC, N, 384, 128, 0, 0);
    k_copy<<<g1(n128), TPB, 0, stream>>>(PEb, XE, n128);
    k_copy<<<g1(n128), TPB, 0, stream>>>(PCb, XC, n128);

    // GINE layers
    for (int l = 0; l < 2; ++l) {
        cvtA(edge_attr, E, 8, 32); cvtW(Leew[l], 8, 128, 32);
        gemm(AH, Leeb[l], EBUF, E, 32, 128, 0, 0);
        bn_stats(EBUF, E); bn_fin(Leebng[l], Leebnb[l], E);
        k_bn_relu_h<<<g1(e128), TPB, 0, stream>>>(EBUF, SCALE, SHIFT, EH, e128);

        // x_ego branch
        k_cvt_h<<<g1(n128), TPB, 0, stream>>>(XE, XH, n128);
        zero(AGG, n128);
        k_edge_msg<<<g1(E * 64), TPB, 0, stream>>>(XH, EH, e_src, e_dst, AGG, E * 64, E);
        k_add_h<<<g1(n128), TPB, 0, stream>>>(XE, AGG, AH, n128);
        cvtW(Lg1a[l], 128, 128, 128);
        gemm(AH, nullptr, BH, N, 128, 128, 1, 1);
        cvtW(Lg1b[l], 128, 128, 128);
        gemm(BH, nullptr, XE, N, 128, 128, 0, 0);

        // x_cut branch
        k_cvt_h<<<g1(n128), TPB, 0, stream>>>(XC, XH, n128);
        zero(AGG, n128);
        k_edge_msg<<<g1(E * 64), TPB, 0, stream>>>(XH, EH, e_src, e_dst, AGG, E * 64, E);
        k_add_h<<<g1(n128), TPB, 0, stream>>>(XC, AGG, AH, n128);
        cvtW(Lg2a[l], 128, 128, 128);
        gemm(AH, nullptr, BH, N, 128, 128, 1, 1);
        cvtW(Lg2b[l], 128, 128, 128);
        gemm(BH, nullptr, XC, N, 128, 128, 0, 0);

        // bn + relu + residual (prev := new)
        bn_stats(XE, N); bn_fin(Lbn0g[l], Lbn0b[l], N);
        k_bn_relu_res<<<g1(n128), TPB, 0, stream>>>(XE, SCALE, SHIFT, PEb, XE, n128);
        bn_stats(XC, N); bn_fin(Lbn1g[l], Lbn1b[l], N);
        k_bn_relu_res<<<g1(n128), TPB, 0, stream>>>(XC, SCALE, SHIFT, PCb, XC, n128);
    }

    // output projections to PE=16
    cvtA(EP, N, 128, 128); cvtW(ego_out_w, 128, 16, 128);
    gemm(AH, ego_out_b, EO, N, 128, 16, 1, 0);
    cvtA(CT, N, 128, 128); cvtW(cut_out_w, 128, 16, 128);
    gemm(AH, cut_out_b, CO, N, 128, 16, 1, 0);
    cvtA(GLO, N, 128, 128); cvtW(glo_out_w, 128, 16, 128);
    gemm(AH, glo_out_b, GO, N, 128, 16, 1, 0);

    // hp = concat[x_ego, glo_o, ego_o] @ whp ; seg_mean over batch
    concatH(160, 0, XE, 128, N);
    concatH(160, 128, GO, 16, N);
    concatH(160, 144, EO, 16, N);
    cvtW(whp_w, 160, 128, 160);
    gemm(AH, whp_b, TMP, N, 160, 128, 0, 0);
    zero(R, G * 128); zero(CNTG, G);
    k_segsum<<<g1(n128), TPB, 0, stream>>>(R, CNTG, TMP, batch, n128, 128);
    k_div_cnt<<<g1(G * 128), TPB, 0, stream>>>(R, CNTG, G * 128, 128);

    // hp2 = concat[x_cut, glo_o, cut_o] @ whp2 ; seg_mean
    concatH(160, 0, XC, 128, N);
    concatH(160, 144, CO, 16, N);
    cvtW(whp2_w, 160, 128, 160);
    gemm(AH, whp2_b, TMP, N, 160, 128, 0, 0);
    zero(RC, G * 128); zero(CNTG, G);
    k_segsum<<<g1(n128), TPB, 0, stream>>>(RC, CNTG, TMP, batch, n128, 128);
    k_div_cnt<<<g1(G * 128), TPB, 0, stream>>>(RC, CNTG, G * 128, 128);

    // decoder
    concatH(256, 0, R, 128, G);
    concatH(256, 128, RC, 128, G);
    cvtW(dec1_w, 256, 128, 256);
    gemm(AH, dec1_b, D1, G, 256, 128, 1, 0);
    k_dec2<<<dim3((unsigned)G), dim3(128), 0, stream>>>(D1, dec2_w, dec2_b, (float*)d_out);
}
